// TDPSOM_11390253269526
// MI455X (gfx1250) — matrix operand order repaired and checked by reading, not compiled, not run on hardware
//
#include <hip/hip_runtime.h>
#include <hip/hip_bf16.h>

// Problem constants (match reference)
#define BATCH 2048
#define SEQT  256
#define DIN   128
#define HID   256
#define PROT  2048

typedef __attribute__((ext_vector_type(16))) __bf16 v16bf;
typedef __attribute__((ext_vector_type(8)))  float  v8f;
typedef int v4i_gv __attribute__((vector_size(16)));   // matches builtin param type

__device__ __forceinline__ v8f wmma_bf16(v16bf a, v16bf b, v8f c) {
    // D(f32,16x16) = A(bf16,16x32) * B(bf16,32x16) + C
    return __builtin_amdgcn_wmma_f32_16x16x32_bf16(
        /*neg_a=*/false, a, /*neg_b=*/false, b,
        /*c_mod=*/(short)0, c, /*reuse_a=*/false, /*reuse_b=*/false);
}

// ---- CDNA5 async global->LDS staging (ASYNCcnt path), with safe fallback ----
#if defined(__gfx1250__) && __has_builtin(__builtin_amdgcn_global_load_async_to_lds_b128)
#define HAVE_ASYNC_LDS 1
#else
#define HAVE_ASYNC_LDS 0
#endif

__device__ __forceinline__ void stage16(const float* __restrict__ g, float* lds) {
#if HAVE_ASYNC_LDS
    __builtin_amdgcn_global_load_async_to_lds_b128(
        (__attribute__((address_space(1))) v4i_gv*)g,
        (__attribute__((address_space(3))) v4i_gv*)lds,
        /*imm offset=*/0, /*cpol=*/0);
#else
    *(float4*)lds = *(const float4*)g;
#endif
}

__device__ __forceinline__ void stage_join() {
#if HAVE_ASYNC_LDS
#if __has_builtin(__builtin_amdgcn_s_wait_asynccnt)
    __builtin_amdgcn_s_wait_asynccnt(0);
#else
    asm volatile("s_wait_asynccnt 0x0" ::: "memory");
#endif
#endif
    __syncthreads();
}

// ---------------------------------------------------------------------------
// Pack a row-major fp32 matrix W[Nrows][Kcols] (used as B^T: B[k][n]=W[n][k])
// into bf16 WMMA B-operand layout: tiles of 32(K)x16(N), 512 elems/tile,
// tile index = nt*kblocks + kt, within tile lane l holds n = l%16 and
// k = kt*32 + (l<16 ? 0 : 16) + j  for j=0..15 (contiguous k order; A uses the same).
// ---------------------------------------------------------------------------
__global__ void pack_b_bf16(const float* __restrict__ W, int Kcols,
                            int ntiles, int kblocks, __bf16* __restrict__ out) {
    int tid = blockIdx.x * blockDim.x + threadIdx.x;
    int total = ntiles * kblocks * 32;
    if (tid >= total) return;
    int lane = tid & 31;
    int tile = tid >> 5;
    int kt = tile % kblocks;
    int nt = tile / kblocks;
    int n = nt * 16 + (lane & 15);
    int kbase = kt * 32 + ((lane & 16) ? 16 : 0);
    const float* src = W + (size_t)n * Kcols + kbase;
    __bf16* dst = out + (size_t)tile * 512 + lane * 16;
#pragma unroll
    for (int j = 0; j < 16; ++j) dst[j] = (__bf16)src[j];
}

__global__ void proto_norms_k(const float* __restrict__ Pm, float* __restrict__ pn) {
    int p = blockIdx.x * blockDim.x + threadIdx.x;
    if (p >= PROT) return;
    const float4* row = (const float4*)(Pm + (size_t)p * HID);
    float s = 0.f;
#pragma unroll 4
    for (int j = 0; j < HID / 4; ++j) {
        float4 v = row[j];
        s += v.x * v.x + v.y * v.y + v.z * v.z + v.w * v.w;
    }
    pn[p] = s;
}

__global__ void zero_f32(float* __restrict__ p, int n) {
    int i = blockIdx.x * blockDim.x + threadIdx.x;
    if (i < n) p[i] = 0.f;
}

// ---------------------------------------------------------------------------
// One GRU timestep, input projection fused.
// Grid: (BATCH/32, HID/32); block: 128 threads = 4 waves.
// Wave w -> (mt = w>>1, nt = w&1) 16x16 output tile inside the 32x32 block
// tile. Each wave accumulates 4 f32 WMMA accumulators: r, z, xn, hn.
// ---------------------------------------------------------------------------
#define XPITCH 132   // 132 % 64 banks -> conflict-free column gathers
#define HPITCH 260

__global__ void __launch_bounds__(128) gru_step_k(
    const float* __restrict__ x, const __bf16* __restrict__ wihp,
    const __bf16* __restrict__ whhp, const float* __restrict__ b_ih,
    const float* __restrict__ b_hh, const float* __restrict__ hprev,
    float* __restrict__ hnext, int t) {
    __shared__ float xs[32 * XPITCH];
    __shared__ float hs[32 * HPITCH];

    const int tid = threadIdx.x;
    const int brow0 = blockIdx.x * 32;

    // Stage x[:, t, :] slice (32 rows x 128) and h_prev (32 rows x 256) to LDS
    // via the CDNA5 async global->LDS DMA path (no VGPR round trip).
    {
        const float* xg = x + (size_t)brow0 * SEQT * DIN + (size_t)t * DIN;
        for (int i = tid; i < 32 * (DIN / 4); i += 128) {
            int r = i >> 5;            // DIN/4 == 32 float4 per row
            int c = (i & 31) * 4;
            stage16(xg + (size_t)r * SEQT * DIN + c, xs + r * XPITCH + c);
        }
        const float* hg = hprev + (size_t)brow0 * HID;
        for (int i = tid; i < 32 * (HID / 4); i += 128) {
            int r = i >> 6;            // HID/4 == 64 float4 per row
            int c = (i & 63) * 4;
            stage16(hg + (size_t)r * HID + c, hs + r * HPITCH + c);
        }
    }
    stage_join();

    const int wave = tid >> 5, lane = tid & 31;
    const int mt = wave >> 1, nt = wave & 1;
    const int n0 = blockIdx.y * 32 + nt * 16;      // output column base in H
    const int lhalf = (lane & 16) ? 16 : 0;        // A-layout K half select
    const int arow = mt * 16 + (lane & 15);        // A-layout row in LDS

    v8f accR  = {0.f,0.f,0.f,0.f,0.f,0.f,0.f,0.f};
    v8f accZ  = accR, accXN = accR, accHN = accR;

    // ---- x @ w_ih^T contribution: K = 128 -> 4 k-blocks of 32 ----
#pragma unroll
    for (int kt = 0; kt < 4; ++kt) {
        v16bf a;
#pragma unroll
        for (int j = 0; j < 16; ++j) {
            int k = kt * 32 + lhalf + j;
            a[j] = (__bf16)xs[arow * XPITCH + k];
        }
#pragma unroll
        for (int g = 0; g < 3; ++g) {
            int ntile = (g * HID + n0) >> 4;       // over 3H rows of w_ih
            v16bf b = *(const v16bf*)(wihp + ((size_t)(ntile * 4 + kt)) * 512 + lane * 16);
            if (g == 0)      accR  = wmma_bf16(a, b, accR);
            else if (g == 1) accZ  = wmma_bf16(a, b, accZ);
            else             accXN = wmma_bf16(a, b, accXN);
        }
    }
    // ---- h @ w_hh^T contribution: K = 256 -> 8 k-blocks of 32 ----
#pragma unroll
    for (int kt = 0; kt < 8; ++kt) {
        v16bf a;
#pragma unroll
        for (int j = 0; j < 16; ++j) {
            int k = kt * 32 + lhalf + j;
            a[j] = (__bf16)hs[arow * HPITCH + k];
        }
#pragma unroll
        for (int g = 0; g < 3; ++g) {
            int ntile = (g * HID + n0) >> 4;       // over 3H rows of w_hh
            v16bf b = *(const v16bf*)(whhp + ((size_t)(ntile * 8 + kt)) * 512 + lane * 16);
            if (g == 0)      accR  = wmma_bf16(a, b, accR);
            else if (g == 1) accZ  = wmma_bf16(a, b, accZ);
            else             accHN = wmma_bf16(a, b, accHN);
        }
    }

    // ---- gate math on C-layout registers ----
    const int ncol = n0 + (lane & 15);             // global column in H
    const float br  = b_ih[ncol]           + b_hh[ncol];
    const float bz  = b_ih[HID + ncol]     + b_hh[HID + ncol];
    const float bxn = b_ih[2 * HID + ncol];
    const float bhn = b_hh[2 * HID + ncol];
    const int mbase = mt * 16 + ((lane & 16) ? 8 : 0);
#pragma unroll
    for (int v = 0; v < 8; ++v) {
        int mloc = mbase + v;
        float hold = hs[mloc * HPITCH + ncol];
        float r  = 1.f / (1.f + __expf(-(accR[v] + br)));
        float z  = 1.f / (1.f + __expf(-(accZ[v] + bz)));
        float nn = tanhf(accXN[v] + bxn + r * (accHN[v] + bhn));
        hnext[(size_t)(brow0 + mloc) * HID + ncol] = (1.f - z) * nn + z * hold;
    }
}

// ---------------------------------------------------------------------------
// argmin_p( ||p||^2 - 2 h.p ) — same argmin as cdist (monotone transform).
// Grid: BATCH/16 blocks; block 128 threads = 4 waves, each wave scans 512
// prototypes (32 n-tiles x 8 k-blocks of WMMA).
// ---------------------------------------------------------------------------
__global__ void __launch_bounds__(128) cdist_argmin_k(
    const float* __restrict__ h, const __bf16* __restrict__ pp,
    const float* __restrict__ pn, int* __restrict__ out) {
    __shared__ float hsm[16 * HPITCH];
    __shared__ float redv[4][16];
    __shared__ int   redi[4][16];

    const int tid = threadIdx.x;
    const int b0 = blockIdx.x * 16;
    for (int i = tid; i < 16 * (HID / 4); i += 128) {
        int r = i >> 6;
        int c = (i & 63) * 4;
        stage16(h + (size_t)(b0 + r) * HID + c, hsm + r * HPITCH + c);
    }
    stage_join();

    const int wave = tid >> 5, lane = tid & 31;
    const int lhalf = (lane & 16) ? 16 : 0;
    const int arow = lane & 15;

    // Hoist all 8 A tiles (shared across every prototype tile this wave does).
    v16bf a8[8];
#pragma unroll
    for (int kt = 0; kt < 8; ++kt) {
#pragma unroll
        for (int j = 0; j < 16; ++j) {
            int k = kt * 32 + lhalf + j;
            a8[kt][j] = (__bf16)hsm[arow * HPITCH + k];
        }
    }

    float mv[8];
    int   mi[8];
#pragma unroll
    for (int v = 0; v < 8; ++v) { mv[v] = 3.4e38f; mi[v] = 0; }

    for (int nt = wave * 32; nt < wave * 32 + 32; ++nt) {
        v8f acc = {0.f,0.f,0.f,0.f,0.f,0.f,0.f,0.f};
#pragma unroll
        for (int kt = 0; kt < 8; ++kt) {
            v16bf b = *(const v16bf*)(pp + ((size_t)(nt * 8 + kt)) * 512 + lane * 16);
            acc = wmma_bf16(a8[kt], b, acc);
        }
        int p = nt * 16 + (lane & 15);
        float pnv = pn[p];
#pragma unroll
        for (int v = 0; v < 8; ++v) {
            float s = pnv - 2.f * acc[v];
            if (s < mv[v]) { mv[v] = s; mi[v] = p; }
        }
    }

    // Reduce across the 16 lanes that share a row (xor offsets < 16 stay in half).
#pragma unroll
    for (int v = 0; v < 8; ++v) {
        for (int off = 1; off < 16; off <<= 1) {
            float ov = __shfl_xor(mv[v], off, 32);
            int   oi = __shfl_xor(mi[v], off, 32);
            if (ov < mv[v]) { mv[v] = ov; mi[v] = oi; }
        }
    }
    if ((lane & 15) == 0) {
        int rbase = (lane & 16) ? 8 : 0;
#pragma unroll
        for (int v = 0; v < 8; ++v) {
            redv[wave][rbase + v] = mv[v];
            redi[wave][rbase + v] = mi[v];
        }
    }
    __syncthreads();
    if (tid < 16) {
        float bv = redv[0][tid];
        int   bi = redi[0][tid];
        for (int w = 1; w < 4; ++w)
            if (redv[w][tid] < bv) { bv = redv[w][tid]; bi = redi[w][tid]; }
        out[b0 + tid] = bi;
    }
}

// ---------------------------------------------------------------------------
extern "C" void kernel_launch(void* const* d_in, const int* in_sizes, int n_in,
                              void* d_out, int out_size, void* d_ws, size_t ws_size,
                              hipStream_t stream) {
    (void)in_sizes; (void)n_in; (void)out_size; (void)ws_size;
    const float* x      = (const float*)d_in[0];   // [B,T,D]
    const float* w_ih   = (const float*)d_in[1];   // [3H,D]
    const float* w_hh   = (const float*)d_in[2];   // [3H,H]
    const float* b_ih   = (const float*)d_in[3];   // [3H]
    const float* b_hh   = (const float*)d_in[4];   // [3H]
    const float* protos = (const float*)d_in[5];   // [P,H]
    int* out = (int*)d_out;                        // [B] argmin indices

    // Workspace layout (~5.6 MB)
    char* ws = (char*)d_ws;
    __bf16* wihp = (__bf16*)ws;                                   // 768*128 bf16
    __bf16* whhp = (__bf16*)(ws + 196608);                        // 768*256 bf16
    __bf16* pp   = (__bf16*)(ws + 196608 + 393216);               // 2048*256 bf16
    float*  pn   = (float*)(ws + 196608 + 393216 + 1048576);      // 2048 f32
    float*  h0   = (float*)(ws + 196608 + 393216 + 1048576 + 8192);
    float*  h1   = h0 + (size_t)BATCH * HID;

    // Pack weights/prototypes into WMMA B layout (bf16).
    {
        int thr = 48 * 4 * 32;   // w_ih: 48 n-tiles, 4 k-blocks
        pack_b_bf16<<<(thr + 255) / 256, 256, 0, stream>>>(w_ih, DIN, 48, 4, wihp);
        thr = 48 * 8 * 32;       // w_hh: 48 n-tiles, 8 k-blocks
        pack_b_bf16<<<(thr + 255) / 256, 256, 0, stream>>>(w_hh, HID, 48, 8, whhp);
        thr = 128 * 8 * 32;      // prototypes: 128 n-tiles, 8 k-blocks
        pack_b_bf16<<<(thr + 255) / 256, 256, 0, stream>>>(protos, HID, 128, 8, pp);
    }
    proto_norms_k<<<PROT / 256, 256, 0, stream>>>(protos, pn);
    zero_f32<<<(BATCH * HID + 255) / 256, 256, 0, stream>>>(h0, BATCH * HID);

    // 256 sequential GRU steps, ping-pong h buffers (final lands in h0).
    for (int t = 0; t < SEQT; ++t) {
        const float* src = (t & 1) ? h1 : h0;
        float*       dst = (t & 1) ? h0 : h1;
        gru_step_k<<<dim3(BATCH / 32, HID / 32), 128, 0, stream>>>(
            x, wihp, whhp, b_ih, b_hh, src, dst, t);
    }

    cdist_argmin_k<<<BATCH / 16, 128, 0, stream>>>(h0, pp, pn, out);
}
